// InstantPolicy_81527069212717
// MI455X (gfx1250) — matrix operand order repaired and checked by reading, not compiled, not run on hardware
//
#include <hip/hip_runtime.h>
#include <hip/hip_bf16.h>

// Problem constants (match reference)
#define NN    131072
#define TT    2
#define DD    512
#define HH    512
#define EPTN  65536
#define MROWS (NN / TT)   // 65536 rows per type

typedef __bf16 bf16x16 __attribute__((ext_vector_type(16)));
typedef __bf16 bf16x8  __attribute__((ext_vector_type(8)));
typedef __bf16 bf16x2  __attribute__((ext_vector_type(2)));
typedef float  f32x8   __attribute__((ext_vector_type(8)));

union Frag16 { bf16x16 v; bf16x8 h[2]; };

// Pack two f32 -> {bf16 lo, bf16 hi} in one dword.  Prefer the native packed
// convert (v_cvt_pk_bf16_f32) when the toolchain declares it; otherwise let
// clang lower scalar __bf16 casts (native bf16 cvt on gfx1250).
__device__ __forceinline__ unsigned pk2bf(float lo, float hi) {
#if __has_builtin(__builtin_amdgcn_cvt_pk_bf16_f32)
  bf16x2 p = __builtin_amdgcn_cvt_pk_bf16_f32(lo, hi);
  return __builtin_bit_cast(unsigned, p);
#else
  bf16x2 p;
  p.x = (__bf16)lo;
  p.y = (__bf16)hi;
  return __builtin_bit_cast(unsigned, p);
#endif
}

constexpr int BM = 128;  // M tile
constexpr int BN = 64;   // N tile
constexpr int BK = 32;   // K tile (WMMA bf16 K)
constexpr int KT = DD / BK;  // 16 K-steps

// ---------------------------------------------------------------------------
// GEMM: Ca = A@Wa + ba (and, if DUAL, Cb = A@Wb + bb sharing the A tile).
// fp32 inputs converted to packed-bf16 in registers; v_wmma_f32_16x16x32_bf16.
// Block = 256 threads = 8 wave32; wave (wm,wn) owns a 32x32 patch (2x2 WMMA
// tiles) of the 128x64 block tile.  Double-buffered LDS software pipeline:
// global loads for tile k+1 overlap the WMMAs on tile k; one barrier/step.
// ---------------------------------------------------------------------------
template <bool DUAL>
__global__ __launch_bounds__(256)
void gemm_bias_wmma(const float* __restrict__ A,
                    const float* __restrict__ Wa, const float* __restrict__ ba,
                    float* __restrict__ Ca,
                    const float* __restrict__ Wb, const float* __restrict__ bb,
                    float* __restrict__ Cb) {
  // Packed bf16-pair tiles (u32 = {K even, K odd}); double buffered.
  __shared__ __align__(16) unsigned As [2][BM][BK / 2];  // 2 x 8 KB
  __shared__ __align__(16) unsigned Bsa[2][BN][BK / 2];  // 2 x 4 KB
  __shared__ __align__(16) unsigned Bsb[2][BN][BK / 2];  // 2 x 4 KB (DUAL)

  const int tid  = threadIdx.x;
  const int m0   = blockIdx.x * BM;
  const int n0   = blockIdx.y * BN;
  const int wid  = tid >> 5;
  const int lane = tid & 31;
  const int wm   = wid >> 1;      // 0..3  -> rows [wm*32, wm*32+32)
  const int wn   = wid & 1;       // 0..1  -> cols [wn*32, wn*32+32)
  const int lr   = lane & 15;
  const int lh   = lane >> 4;

  // Staging assignment.
  const int ar  = tid >> 1;        // A row 0..127
  const int ac  = (tid & 1) * 16;  // A col 0 or 16
  const int bkp = tid >> 4;        // B k-pair 0..15 (rows 2*bkp, 2*bkp+1)
  const int bn  = (tid & 15) * 4;  // B col 0..60

  const float* Ap  = A  + (size_t)(m0 + ar) * DD + ac;
  const float* Wap = Wa + (size_t)(2 * bkp) * HH + n0 + bn;
  const float* Wbp = DUAL ? (Wb + (size_t)(2 * bkp) * HH + n0 + bn) : nullptr;

  float4 ra[4], wa0, wa1, wb0, wb1;

  auto load_tile = [&](int k0) {
    const float4* pa = (const float4*)(Ap + k0);
#pragma unroll
    for (int i = 0; i < 4; ++i) ra[i] = pa[i];
    const float* pw = Wap + (size_t)k0 * HH;
    wa0 = *(const float4*)pw;
    wa1 = *(const float4*)(pw + HH);
    if constexpr (DUAL) {
      const float* qw = Wbp + (size_t)k0 * HH;
      wb0 = *(const float4*)qw;
      wb1 = *(const float4*)(qw + HH);
    }
  };
  auto store_tile = [&](int buf) {
    uint4 p0, p1;
    p0.x = pk2bf(ra[0].x, ra[0].y); p0.y = pk2bf(ra[0].z, ra[0].w);
    p0.z = pk2bf(ra[1].x, ra[1].y); p0.w = pk2bf(ra[1].z, ra[1].w);
    p1.x = pk2bf(ra[2].x, ra[2].y); p1.y = pk2bf(ra[2].z, ra[2].w);
    p1.z = pk2bf(ra[3].x, ra[3].y); p1.w = pk2bf(ra[3].z, ra[3].w);
    *(uint4*)&As[buf][ar][(ac >> 1) + 0] = p0;   // ds_store_b128
    *(uint4*)&As[buf][ar][(ac >> 1) + 4] = p1;   // ds_store_b128
    // B stored (N,K/2): word = {K=2kp, K=2kp+1} for column n.
    Bsa[buf][bn + 0][bkp] = pk2bf(wa0.x, wa1.x);
    Bsa[buf][bn + 1][bkp] = pk2bf(wa0.y, wa1.y);
    Bsa[buf][bn + 2][bkp] = pk2bf(wa0.z, wa1.z);
    Bsa[buf][bn + 3][bkp] = pk2bf(wa0.w, wa1.w);
    if constexpr (DUAL) {
      Bsb[buf][bn + 0][bkp] = pk2bf(wb0.x, wb1.x);
      Bsb[buf][bn + 1][bkp] = pk2bf(wb0.y, wb1.y);
      Bsb[buf][bn + 2][bkp] = pk2bf(wb0.z, wb1.z);
      Bsb[buf][bn + 3][bkp] = pk2bf(wb0.w, wb1.w);
    }
  };

  f32x8 acc_a[2][2] = {};
  f32x8 acc_b[2][2] = {};

  // Prologue: stage tile 0.
  load_tile(0);
  store_tile(0);
  __syncthreads();

  for (int kt = 0; kt < KT; ++kt) {
    const int cur = kt & 1;
    if (kt + 1 < KT) {
      load_tile((kt + 1) * BK);                       // overlap with WMMA below
      if (kt + 2 < KT)
        __builtin_prefetch(Ap + (kt + 2) * BK, 0, 1); // global_prefetch_b8
    }

    // A fragments: contiguous k order, same as B (lane<16: K0..15;
    // lane>=16: K16..31); row M = lane%16.
    Frag16 fa[2];
#pragma unroll
    for (int mi = 0; mi < 2; ++mi) {
      const int row = wm * 32 + mi * 16 + lr;
      fa[mi].h[0] = *(const bf16x8*)&As[cur][row][lh * 8];
      fa[mi].h[1] = *(const bf16x8*)&As[cur][row][lh * 8 + 4];
    }
    // B fragments: 32x16, col = lane%16; lane<16 K0..15, lane>=16 K16..31.
    Frag16 fba[2], fbb[2];
#pragma unroll
    for (int ni = 0; ni < 2; ++ni) {
      const int col = wn * 32 + ni * 16 + lr;
      fba[ni].h[0] = *(const bf16x8*)&Bsa[cur][col][lh * 8];
      fba[ni].h[1] = *(const bf16x8*)&Bsa[cur][col][lh * 8 + 4];
      if constexpr (DUAL) {
        fbb[ni].h[0] = *(const bf16x8*)&Bsb[cur][col][lh * 8];
        fbb[ni].h[1] = *(const bf16x8*)&Bsb[cur][col][lh * 8 + 4];
      }
    }
#pragma unroll
    for (int mi = 0; mi < 2; ++mi)
#pragma unroll
      for (int ni = 0; ni < 2; ++ni) {
        acc_a[mi][ni] = __builtin_amdgcn_wmma_f32_16x16x32_bf16(
            false, fa[mi].v, false, fba[ni].v, (short)0, acc_a[mi][ni], false, false);
        if constexpr (DUAL)
          acc_b[mi][ni] = __builtin_amdgcn_wmma_f32_16x16x32_bf16(
              false, fa[mi].v, false, fbb[ni].v, (short)0, acc_b[mi][ni], false, false);
      }

    if (kt + 1 < KT) store_tile(cur ^ 1);
    __syncthreads();
  }

  // Epilogue: C/D 16x16 f32 layout: VGPR j -> M = j + (lane>=16)*8.
#pragma unroll
  for (int mi = 0; mi < 2; ++mi)
#pragma unroll
    for (int ni = 0; ni < 2; ++ni) {
      const int col_g = n0 + wn * 32 + ni * 16 + lr;
      const float bva = ba[col_g];
      const float bvb = DUAL ? bb[col_g] : 0.0f;
#pragma unroll
      for (int j = 0; j < 8; ++j) {
        const int m = wm * 32 + mi * 16 + j + lh * 8;
        Ca[(size_t)(m0 + m) * HH + col_g] = acc_a[mi][ni][j] + bva;
        if constexpr (DUAL)
          Cb[(size_t)(m0 + m) * HH + col_g] = acc_b[mi][ni][j] + bvb;
      }
    }
}

// ---------------------------------------------------------------------------
// Edge scatter: out[src] += w2f[dst] + w5f[e].  (softmax over a length-1 axis
// is identically 1.0, so the attention score is exactly 1 — w3f/w4f are dead.)
// One block (128 threads) per edge, float4 gathers, hardware fp32 atomics.
// ---------------------------------------------------------------------------
__global__ __launch_bounds__(128)
void edge_scatter(const long long* __restrict__ edges,
                  const float* __restrict__ w2f,
                  const float* __restrict__ w5f,
                  float* __restrict__ out) {
  const size_t e = blockIdx.x;
  const long long src = edges[2 * e + 0];   // uniform -> SMEM load
  const long long dst = edges[2 * e + 1];
  const int h = threadIdx.x * 4;
  const float4 a = *(const float4*)(w2f + (size_t)dst * HH + h);
  const float4 b = *(const float4*)(w5f + e * HH + h);
  float* o = out + (size_t)src * HH + h;
  unsafeAtomicAdd(o + 0, a.x + b.x);   // global_atomic_add_f32
  unsafeAtomicAdd(o + 1, a.y + b.y);
  unsafeAtomicAdd(o + 2, a.z + b.z);
  unsafeAtomicAdd(o + 3, a.w + b.w);
}

extern "C" void kernel_launch(void* const* d_in, const int* in_sizes, int n_in,
                              void* d_out, int out_size, void* d_ws, size_t ws_size,
                              hipStream_t stream) {
  const float* X  = (const float*)d_in[0];
  const float* E0 = (const float*)d_in[1];
  const float* E1 = (const float*)d_in[2];
  const float* W1 = (const float*)d_in[3];
  const float* b1 = (const float*)d_in[4];
  const float* W2 = (const float*)d_in[5];
  const float* b2 = (const float*)d_in[6];
  // W3/b3 (7,8) and W4/b4 (9,10) are mathematically dead: softmax over a
  // singleton axis == 1.0, so score*(w2f[dst]+w5f) == w2f[dst]+w5f exactly.
  const float* W5 = (const float*)d_in[11];
  const float* b5 = (const float*)d_in[12];
  const long long* edges0 = (const long long*)d_in[13];
  const long long* edges1 = (const long long*)d_in[14];
  float* out = (float*)d_out;

  // Workspace: w2f (N*H) | w5f_0 (EPT*H) | w5f_1 (EPT*H)  = 536.9 MB fp32
  float* w2f  = (float*)d_ws;
  float* w5f0 = w2f + (size_t)NN * HH;
  float* w5f1 = w5f0 + (size_t)EPTN * HH;

  dim3 grid(MROWS / BM, HH / BN);   // (512, 8)
  dim3 blk(256);

  // w1f -> out (direct), w2f -> workspace.  X read once per type (fused).
  for (int t = 0; t < TT; ++t) {
    const size_t xo = (size_t)t * MROWS * DD;
    const size_t wo = (size_t)t * DD * HH;
    const size_t bo = (size_t)t * HH;
    gemm_bias_wmma<true><<<grid, blk, 0, stream>>>(
        X + xo, W1 + wo, b1 + bo, out + xo, W2 + wo, b2 + bo, w2f + xo);
  }
  // w5f for both edge types
  gemm_bias_wmma<false><<<grid, blk, 0, stream>>>(
      E0, W5, b5, w5f0, nullptr, nullptr, nullptr);
  gemm_bias_wmma<false><<<grid, blk, 0, stream>>>(
      E1, W5 + (size_t)DD * HH, b5 + HH, w5f1, nullptr, nullptr, nullptr);

  // Scatter-add contributions
  edge_scatter<<<EPTN, 128, 0, stream>>>(edges0, w2f, w5f0, out);
  edge_scatter<<<EPTN, 128, 0, stream>>>(edges1, w2f, w5f1, out);
}
